// LSTM_55147380081132
// MI455X (gfx1250) — matrix operand order repaired and checked by reading, not compiled, not run on hardware
//
#include <hip/hip_runtime.h>

#define H_     1024
#define G4     4096
#define BATCH  32
#define SEQ    512
#define VOCAB  256

#define L0_NB  64
#define L1_NB  64
#define Y_NB   2
#define NWG    (L0_NB + L1_NB + Y_NB)   // 130 workgroups
#define TPB    128                      // 4 waves (wave32)

// dynamic LDS: h0 stage (64KB) + h1 stage (64KB) + gate exchange (8KB)
#define SMEM_BYTES (2 * BATCH * H_ * 2 + 4 * 32 * 16 * 4)

typedef __attribute__((ext_vector_type(16))) __bf16       v16bf;
typedef __attribute__((ext_vector_type(8)))  float        v8f;
typedef __attribute__((ext_vector_type(4)))  unsigned int u32x4;
typedef unsigned short u16;

union FragU { u32x4 u[2]; v16bf v; };

static __device__ __forceinline__ u16 f2bf(float f) {
    unsigned u = __float_as_uint(f);
    u += 0x7FFFu + ((u >> 16) & 1u);          // round-to-nearest-even
    return (u16)(u >> 16);
}

static __device__ __forceinline__ v8f splat8(float v) {
    v8f r;
#pragma unroll
    for (int i = 0; i < 8; ++i) r[i] = v;
    return r;
}

static __device__ __forceinline__ float sigm(float v) {
    return 1.0f / (1.0f + __expf(-v));
}

// A fragment: 16x32 bf16, rows m0..m0+15 of a 32xH row-major bf16 matrix.
// lane<16 holds K {0..7,16..23}, lane>=16 holds K {8..15,24..31} (ISA 7.12.2).
static __device__ __forceinline__ v16bf load_afrag(const u16* hbase, int m0, int k0, int lane) {
    const u16* p = hbase + (size_t)(m0 + (lane & 15)) * H_ + k0 + ((lane & 16) ? 8 : 0);
    FragU r;
    r.u[0] = *(const u32x4*)p;        // K base .. base+7
    r.u[1] = *(const u32x4*)(p + 16); // K base+16 .. base+23
    return r.v;
}

// B fragment: 32x16 bf16 from a column-major (N x K) bf16 weight tile.
// lane<16 holds K {0..7,16..23} of column (lane&15); lane>=16 holds K {8..15,24..31}.
static __device__ __forceinline__ v16bf load_bfrag(const u16* wrow, int k0, int lane) {
    const u16* p = wrow + (size_t)(lane & 15) * H_ + k0 + ((lane & 16) ? 8 : 0);
    FragU r;
    r.u[0] = *(const u32x4*)p;        // K base .. base+7
    r.u[1] = *(const u32x4*)(p + 16); // K base+16 .. base+23
    return r.v;
}

static __device__ __forceinline__ v8f wmma_bf16(v16bf a, v16bf b, v8f c) {
    return __builtin_amdgcn_wmma_f32_16x16x32_bf16(false, a, false, b, (short)0, c, false, false);
}

static __device__ __forceinline__ void gridbar(unsigned* cnt, unsigned target) {
    __builtin_amdgcn_fence(__ATOMIC_RELEASE, "agent");
    __syncthreads();
    if (threadIdx.x == 0) {
        __hip_atomic_fetch_add(cnt, 1u, __ATOMIC_RELAXED, __HIP_MEMORY_SCOPE_AGENT);
        while (__hip_atomic_load(cnt, __ATOMIC_RELAXED, __HIP_MEMORY_SCOPE_AGENT) < target)
            __builtin_amdgcn_s_sleep(8);
    }
    __syncthreads();
    __builtin_amdgcn_fence(__ATOMIC_ACQUIRE, "agent");
}

// LSTM cell elementwise update for one 16-unit group (32 batch rows).
// gst layout: [gate][row(32)][n(16)] f32. Writes bf16 h slice, keeps c in regs.
static __device__ __forceinline__ void cell_update(const float* gst, v8f& cLo, v8f& cHi,
                                                   u16* hdst, int grp, int lane) {
    const int n = lane & 15, half = lane >> 4;
#pragma unroll
    for (int r = 0; r < 8; ++r) {
        int row = r + 8 * half;
        {
            float iv = sigm(gst[(0 * 32 + row) * 16 + n]);
            float fv = sigm(gst[(1 * 32 + row) * 16 + n]);
            float gv = tanhf(gst[(2 * 32 + row) * 16 + n]);
            float ov = sigm(gst[(3 * 32 + row) * 16 + n]);
            float c  = fv * cLo[r] + iv * gv;
            cLo[r]   = c;
            hdst[(size_t)row * H_ + grp * 16 + n] = f2bf(ov * tanhf(c));
        }
        {
            int row2 = row + 16;
            float iv = sigm(gst[(0 * 32 + row2) * 16 + n]);
            float fv = sigm(gst[(1 * 32 + row2) * 16 + n]);
            float gv = tanhf(gst[(2 * 32 + row2) * 16 + n]);
            float ov = sigm(gst[(3 * 32 + row2) * 16 + n]);
            float c  = fv * cHi[r] + iv * gv;
            cHi[r]   = c;
            hdst[(size_t)row2 * H_ + grp * 16 + n] = f2bf(ov * tanhf(c));
        }
    }
}

// One-time prep: convert f32 weights to bf16, transposed to column-major (N x K).
__global__ void lstm_prep(const float* __restrict__ Wh0, const float* __restrict__ Wx1,
                          const float* __restrict__ Wh1, const float* __restrict__ Why,
                          u16* __restrict__ wt) {
    const int m = blockIdx.y;
    const float* src = (m == 0) ? Wh0 : (m == 1) ? Wx1 : (m == 2) ? Wh1 : Why;
    const int N = (m == 3) ? VOCAB : G4;
    u16* dst = wt + (size_t)m * G4 * H_;
    const size_t total = (size_t)N * H_;
    for (size_t e = (size_t)blockIdx.x * blockDim.x + threadIdx.x; e < total;
         e += (size_t)gridDim.x * blockDim.x) {
        size_t n = e / H_, k = e % H_;
        dst[n * H_ + k] = f2bf(src[k * (size_t)N + n]);
    }
}

__global__ void __launch_bounds__(TPB, 1)
lstm_persist(const int* __restrict__ x, const float* __restrict__ Wx0,
             const float* __restrict__ b0, const float* __restrict__ b1,
             const float* __restrict__ by, const u16* __restrict__ wt,
             u16* __restrict__ h0buf, u16* __restrict__ h1buf,
             unsigned* __restrict__ cnt, float* __restrict__ out) {
    extern __shared__ char smem[];
    u16*   hs0 = (u16*)smem;                       // staged h0 (32x1024 bf16)
    u16*   hs1 = hs0 + BATCH * H_;                 // staged h1
    float* gst = (float*)(smem + 2 * BATCH * H_ * 2);  // [4][32][16] gate exchange

    const u16* Wh0t = wt;
    const u16* Wx1t = wt + (size_t)1 * G4 * H_;
    const u16* Wh1t = wt + (size_t)2 * G4 * H_;
    const u16* Whyt = wt + (size_t)3 * G4 * H_;

    const int tid  = threadIdx.x;
    const int lane = tid & 31;
    const int wv   = tid >> 5;                     // wave id == gate id (roles 0/1)
    const int blk  = blockIdx.x;
    const int role = (blk < L0_NB) ? 0 : (blk < L0_NB + L1_NB) ? 1 : 2;
    const int grp  = (role == 0) ? blk : blk - L0_NB;   // 16-unit group (roles 0/1)
    const int n    = lane & 15;
    const int half = lane >> 4;

    v8f cLo = splat8(0.0f), cHi = splat8(0.0f);    // cell state (wave 0, roles 0/1)

    const int gcol   = wv * H_ + grp * 16 + n;     // gate column in [0,4096)
    const float b0v  = (role == 0) ? b0[gcol] : 0.0f;
    const float b1v  = (role == 1) ? b1[gcol] : 0.0f;

    unsigned bar = 0;
    for (int t = 0; t < SEQ; ++t) {
        const int rb = t & 1, wb = rb ^ 1;         // h(t-1) in buf[rb], h(t) -> buf[wb]

        if (role == 0) {
            // ---- layer 0: gates = xproj + h0(t-1) @ W_h0 ----
            const u16* src = h0buf + (size_t)rb * BATCH * H_;
            for (int e = tid; e < BATCH * H_ / 8; e += TPB)
                ((u32x4*)hs0)[e] = ((const u32x4*)src)[e];
            __syncthreads();

            v8f a0, a1;   // rows 0-15, 16-31 of 32x16 gate tile
#pragma unroll
            for (int r = 0; r < 8; ++r) {          // one-hot gather seeds accumulators
                int row = r + 8 * half;
                int xi0 = x[row * SEQ + t];
                int xi1 = x[(row + 16) * SEQ + t];
                a0[r] = Wx0[(size_t)xi0 * G4 + gcol] + b0v;
                a1[r] = Wx0[(size_t)xi1 * G4 + gcol] + b0v;
            }
            const u16* wrow = Wh0t + (size_t)(wv * H_ + grp * 16) * H_;
            for (int k0 = 0; k0 < H_; k0 += 32) {
                v16bf bfr = load_bfrag(wrow, k0, lane);
                v16bf af0 = load_afrag(hs0, 0,  k0, lane);
                v16bf af1 = load_afrag(hs0, 16, k0, lane);
                a0 = wmma_bf16(af0, bfr, a0);
                a1 = wmma_bf16(af1, bfr, a1);
            }
#pragma unroll
            for (int r = 0; r < 8; ++r) {
                int row = r + 8 * half;
                gst[(wv * 32 + row) * 16 + n]      = a0[r];
                gst[(wv * 32 + row + 16) * 16 + n] = a1[r];
            }
            __syncthreads();
            if (wv == 0)
                cell_update(gst, cLo, cHi, h0buf + (size_t)wb * BATCH * H_, grp, lane);
        }
        bar++; gridbar(cnt, bar * NWG);            // h0(t) visible

        if (role == 1) {
            // ---- layer 1: gates = h0(t) @ W_x1 + h1(t-1) @ W_h1 + b1 ----
            const u16* s0 = h0buf + (size_t)wb * BATCH * H_;
            const u16* s1 = h1buf + (size_t)rb * BATCH * H_;
            for (int e = tid; e < BATCH * H_ / 8; e += TPB) {
                ((u32x4*)hs0)[e] = ((const u32x4*)s0)[e];
                ((u32x4*)hs1)[e] = ((const u32x4*)s1)[e];
            }
            __syncthreads();

            v8f a0 = splat8(b1v), a1 = splat8(b1v);
            const u16* wrX = Wx1t + (size_t)(wv * H_ + grp * 16) * H_;
            const u16* wrH = Wh1t + (size_t)(wv * H_ + grp * 16) * H_;
            for (int k0 = 0; k0 < H_; k0 += 32) {
                v16bf bX  = load_bfrag(wrX, k0, lane);
                a0 = wmma_bf16(load_afrag(hs0, 0,  k0, lane), bX, a0);
                a1 = wmma_bf16(load_afrag(hs0, 16, k0, lane), bX, a1);
                v16bf bH  = load_bfrag(wrH, k0, lane);
                a0 = wmma_bf16(load_afrag(hs1, 0,  k0, lane), bH, a0);
                a1 = wmma_bf16(load_afrag(hs1, 16, k0, lane), bH, a1);
            }
#pragma unroll
            for (int r = 0; r < 8; ++r) {
                int row = r + 8 * half;
                gst[(wv * 32 + row) * 16 + n]      = a0[r];
                gst[(wv * 32 + row + 16) * 16 + n] = a1[r];
            }
            __syncthreads();
            if (wv == 0)
                cell_update(gst, cLo, cHi, h1buf + (size_t)wb * BATCH * H_, grp, lane);
        }
        bar++; gridbar(cnt, bar * NWG);            // h1(t) visible

        if (role == 2) {
            // ---- out[t] = h1(t) @ W_hy + b_y  (2 blocks x 4 waves x 2 tiles) ----
            const u16* s1 = h1buf + (size_t)wb * BATCH * H_;
            const int blkoff = (blk - L0_NB - L1_NB) * 128;
#pragma unroll
            for (int sub = 0; sub < 2; ++sub) {
                const int col0 = blkoff + wv * 32 + sub * 16;
                v8f a0 = splat8(by[col0 + n]), a1 = splat8(by[col0 + n]);
                const u16* wrow = Whyt + (size_t)col0 * H_;
                for (int k0 = 0; k0 < H_; k0 += 32) {
                    v16bf bfr = load_bfrag(wrow, k0, lane);
                    a0 = wmma_bf16(load_afrag(s1, 0,  k0, lane), bfr, a0);
                    a1 = wmma_bf16(load_afrag(s1, 16, k0, lane), bfr, a1);
                }
#pragma unroll
                for (int r = 0; r < 8; ++r) {
                    int row = r + 8 * half;
                    out[((size_t)t * BATCH + row) * VOCAB + col0 + n]        = a0[r];
                    out[((size_t)t * BATCH + row + 16) * VOCAB + col0 + n]   = a1[r];
                }
            }
        }
    }
}

extern "C" void kernel_launch(void* const* d_in, const int* in_sizes, int n_in,
                              void* d_out, int out_size, void* d_ws, size_t ws_size,
                              hipStream_t stream) {
    const int*   x   = (const int*)d_in[0];
    const float* Wx0 = (const float*)d_in[1];
    const float* Wh0 = (const float*)d_in[2];
    const float* b0  = (const float*)d_in[3];
    const float* Wx1 = (const float*)d_in[4];
    const float* Wh1 = (const float*)d_in[5];
    const float* b1  = (const float*)d_in[6];
    const float* Why = (const float*)d_in[7];
    const float* by  = (const float*)d_in[8];
    float* out = (float*)d_out;

    char* ws = (char*)d_ws;
    unsigned* cnt = (unsigned*)ws;                                   // barrier counter
    u16* h0b = (u16*)(ws + 256);                                     // [2][32][1024] bf16
    u16* h1b = h0b + 2 * BATCH * H_;
    u16* wt  = (u16*)(ws + 256 + (size_t)4 * BATCH * H_ * sizeof(u16));

    const size_t need = 256 + (size_t)4 * BATCH * H_ * 2 +
                        ((size_t)3 * G4 * H_ + (size_t)VOCAB * H_) * 2;
    if (ws_size < need) return;

    // zero barrier counter + both h ping-pong buffers (h(-1) = 0) each launch
    hipMemsetAsync(d_ws, 0, 256 + (size_t)4 * BATCH * H_ * sizeof(u16), stream);

    lstm_prep<<<dim3(512, 4), 256, 0, stream>>>(Wh0, Wx1, Wh1, Why, wt);

    hipFuncSetAttribute((const void*)lstm_persist,
                        hipFuncAttributeMaxDynamicSharedMemorySize, SMEM_BYTES);
    lstm_persist<<<NWG, TPB, SMEM_BYTES, stream>>>(x, Wx0, b0, b1, by, wt,
                                                   h0b, h1b, cnt, out);
    (void)in_sizes; (void)n_in; (void)out_size;
}
